// GATrNet_81458349736002
// MI455X (gfx1250) — matrix operand order repaired and checked by reading, not compiled, not run on hardware
//
#include <hip/hip_runtime.h>
#include <hip/hip_bf16.h>

typedef __attribute__((ext_vector_type(16))) _Float16 v16h;
typedef __attribute__((ext_vector_type(8)))  float    v8f;

#define BB 8
#define NN 2048
#define CC 16          // channels
#define HH 4           // heads
#define DD 64          // head dim = (C/H)*16
#define TOKENS (BB*NN)

#if defined(__has_builtin)
#  if __has_builtin(__builtin_amdgcn_global_load_async_to_lds_b128)
#    define HAVE_ASYNC_LDS 1
#  endif
#  if __has_builtin(__builtin_amdgcn_tensor_load_to_lds)
#    define HAVE_TDM 1
#  endif
#endif

// builtin expects pointers to gcc-style int4 vectors (per hipcc diagnostic)
typedef int v4i __attribute__((vector_size(16)));
typedef __attribute__((address_space(1))) v4i GV4;
typedef __attribute__((address_space(3))) v4i LV4;
__device__ __forceinline__ GV4* as_global(const void* p) {
    return (GV4*)(unsigned long long)(uintptr_t)p;
}
__device__ __forceinline__ LV4* as_lds(const void* p) {
    return (LV4*)(unsigned int)(uintptr_t)p;   // LDS offset lives in low 32 bits
}

// TDM descriptor vector types (this toolchain: 6-arg tensor_load_to_lds)
typedef unsigned int u32x4 __attribute__((vector_size(16)));
typedef int          i32x8 __attribute__((vector_size(32)));
typedef int          i32x4 __attribute__((vector_size(16)));

// blade tables: blades sorted by (grade, value)
__constant__ int GR[16]   = {0,1,1,1,1,2,2,2,2,2,2,3,3,3,3,4};
__constant__ int MASK[16] = {0,1,2,4,8,3,5,6,9,10,12,7,11,13,14,15};
__constant__ int IDXOF[16]= {0,1,2,5,3,6,7,11,4,8,9,12,10,13,14,15};

__device__ __forceinline__ int gp_sign(int a, int b) {
    int cnt = 0, aa = a >> 1;
    while (aa) { cnt += __popc(aa & b); aa >>= 1; }
    return (cnt & 1) ? -1 : 1;
}

// ---------------- lift: x = clin(stack(p_mv, v_mv), lift_w) ----------------
__global__ void lift_kernel(const float* __restrict__ pos, const float* __restrict__ vel,
                            const float* __restrict__ lw, float* __restrict__ x) {
    int idx = blockIdx.x * blockDim.x + threadIdx.x;
    if (idx >= TOKENS * CC) return;
    int t = idx >> 4, o = idx & 15;
    float l0 = lw[(o*2 + 0)*5 + 1];   // grade-1 weight, input 0 (pos mv)
    float l1 = lw[(o*2 + 1)*5 + 1];   // grade-1 weight, input 1 (vel mv)
    float* xo = x + (size_t)t*256 + o*16;
    float p0 = pos[t*3+0], p1 = pos[t*3+1], p2 = pos[t*3+2];
    float v0 = vel[t*3+0], v1 = vel[t*3+1], v2 = vel[t*3+2];
    xo[0] = 0.f;
    xo[1] = l0;                       // e0 component of p_mv is 1
    xo[2] = p0*l0 + v0*l1;
    xo[3] = p1*l0 + v1*l1;
    xo[4] = p2*l0 + v2*l1;
    #pragma unroll
    for (int b = 5; b < 16; ++b) xo[b] = 0.f;
}

// ---------------- qkv: h = mvnorm(x); q/k/v = clin(h, w*) as f16 ----------------
__global__ void qkv_kernel(const float* __restrict__ x,
                           const float* __restrict__ wq, const float* __restrict__ wk,
                           const float* __restrict__ wv,
                           _Float16* __restrict__ q, _Float16* __restrict__ k,
                           _Float16* __restrict__ v) {
    __shared__ float red[256];
    __shared__ float hs[256];
    int tok = blockIdx.x, t = threadIdx.x;
    int b = tok / NN, n = tok % NN;
    float xv = x[(size_t)tok*256 + t];
    red[t] = xv * xv;
    __syncthreads();
    for (int off = 128; off; off >>= 1) { if (t < off) red[t] += red[t+off]; __syncthreads(); }
    float s = red[0] * (1.0f/16.0f);                 // mean over channels of blade-sums
    hs[t] = xv * rsqrtf(s + 1e-6f);
    __syncthreads();
    int o = t >> 4, bl = t & 15, gr = GR[bl];
    float aq = 0.f, ak = 0.f, av = 0.f;
    #pragma unroll
    for (int i = 0; i < 16; ++i) {
        float h = hs[i*16 + bl];
        aq += h * wq[(o*16+i)*5 + gr];
        ak += h * wk[(o*16+i)*5 + gr];
        av += h * wv[(o*16+i)*5 + gr];
    }
    size_t dst = (((size_t)(b*HH + (o>>2))*NN) + n)*DD + (o&3)*16 + bl;
    q[dst] = (_Float16)aq; k[dst] = (_Float16)ak; v[dst] = (_Float16)av;
}

// ---------------- WMMA fragment gathers ----------------
union H8 { uint4 u; _Float16 h[8]; };

// A operand (16x32 f16): lanes 0-15 row M=lane, K {d..d+7, d+16..d+23}; lanes 16-31 M=lane-16, +8
__device__ __forceinline__ v16h fragA(const _Float16* tile, int rowstride, int lane, int dbase) {
    int row = lane & 15;
    int kb  = dbase + ((lane & 16) ? 8 : 0);
    const _Float16* p = tile + row*rowstride + kb;
    H8 a, b; a.u = *(const uint4*)p; b.u = *(const uint4*)(p + 16);
    v16h r;
    #pragma unroll
    for (int i = 0; i < 8; ++i) { r[i] = a.h[i]; r[i+8] = b.h[i]; }
    return r;
}

// B operand (32x16 f16): lane = col N; lanes 0-15 hold K {d..d+7, d+16..d+23}; lanes 16-31 +8
__device__ __forceinline__ v16h fragB(const _Float16* tile, int rowstride, int lane, int dbase) {
    int col = lane & 15;
    int kb  = dbase + ((lane & 16) ? 8 : 0);
    const _Float16* p = tile + col*rowstride + kb;
    H8 a, b; a.u = *(const uint4*)p; b.u = *(const uint4*)(p + 16);
    v16h r;
    #pragma unroll
    for (int i = 0; i < 8; ++i) { r[i] = a.h[i]; r[i+8] = b.h[i]; }
    return r;
}

// ---- VALU-pipe 16-lane reductions via DPP row rotations (wave32: row = 16 lanes) ----
template <int CTRL>
__device__ __forceinline__ float dpp_rot(float x) {
    return __int_as_float(__builtin_amdgcn_update_dpp(0, __float_as_int(x), CTRL, 0xF, 0xF, true));
}
__device__ __forceinline__ float rowred_max(float v) {
    v = fmaxf(v, dpp_rot<0x121>(v));   // row_ror:1
    v = fmaxf(v, dpp_rot<0x122>(v));   // row_ror:2
    v = fmaxf(v, dpp_rot<0x124>(v));   // row_ror:4
    v = fmaxf(v, dpp_rot<0x128>(v));   // row_ror:8
    return v;
}
__device__ __forceinline__ float rowred_sum(float v) {
    v += dpp_rot<0x121>(v);
    v += dpp_rot<0x122>(v);
    v += dpp_rot<0x124>(v);
    v += dpp_rot<0x128>(v);
    return v;
}

// ---------------- flash attention: one block = 4 waves, one (b,h); wave = 16-query tile ----------------
__global__ void attn_kernel(const _Float16* __restrict__ q, const _Float16* __restrict__ kk,
                            const _Float16* __restrict__ vv, float* __restrict__ obuf) {
    __shared__ alignas(16) _Float16 kbuf[32*DD];      // K chunk row-major (TDM/async-staged)
    __shared__ alignas(16) _Float16 vbufT[DD*32];     // V chunk transposed: [d][key]
    __shared__ alignas(16) _Float16 pbuf[4][16*32];   // per-wave P tile
    int bh   = blockIdx.x;                            // b*H + head
    int lane = threadIdx.x & 31;
    int wave = threadIdx.x >> 5;
    int q0   = blockIdx.y*64 + wave*16;
    const _Float16* qb = q  + (size_t)bh*NN*DD;
    const _Float16* kb = kk + (size_t)bh*NN*DD;
    const _Float16* vb = vv + (size_t)bh*NN*DD;

    v16h aq0 = fragA(qb + (size_t)q0*DD, DD, lane, 0);
    v16h aq1 = fragA(qb + (size_t)q0*DD, DD, lane, 32);

    v8f acc[4] = {};
    float mrow[8], lrow[8];
    #pragma unroll
    for (int r = 0; r < 8; ++r) { mrow[r] = -1e30f; lrow[r] = 0.f; }

    for (int j = 0; j < NN/32; ++j) {
        // ---- stage K chunk (32 keys x 64 dims, 2D tile, 4KB) into LDS ----
#if defined(HAVE_TDM)
        if (wave == 0) {
            unsigned long long ga = (unsigned long long)(uintptr_t)(kb + (size_t)(j*32)*DD);
            unsigned int lds_off  = (unsigned int)(uintptr_t)kbuf;
            // D# group0: count=1 | lds_addr | global_addr(57b) | type=2
            u32x4 g0 = { 1u,
                         lds_off,
                         (unsigned int)ga,
                         (unsigned int)((ga >> 32) & 0x1FFFFFFull) | (2u << 30) };
            // D# group1: data_size=2B; tensor_dim0=64, tensor_dim1=2048, tile=64x32, dim0_stride=64
            i32x8 g1 = { (int)(1u << 16),      // data_size=1 (2 bytes)
                         (int)(64u << 16),     // tensor_dim0[15:0] in [63:48]
                         (int)(2048u << 16),   // tensor_dim0 hi=0 | tensor_dim1[15:0]
                         (int)(64u << 16),     // tensor_dim1 hi=0 | tile_dim0=64
                         32,                   // tile_dim1=32 | tile_dim2=0
                         64,                   // tensor_dim0_stride[31:0]=64
                         0, 0 };               // stride hi / dim1_stride
            i32x4 gz4 = { 0, 0, 0, 0 };
            i32x8 gz8 = { 0, 0, 0, 0, 0, 0, 0, 0 };
            __builtin_amdgcn_tensor_load_to_lds(g0, g1, gz4, gz4, gz8, 0);
#  if __has_builtin(__builtin_amdgcn_s_wait_tensorcnt)
            __builtin_amdgcn_s_wait_tensorcnt(0);
#  else
            asm volatile("s_wait_tensorcnt 0x0" ::: "memory");
#  endif
        }
#else
        #pragma unroll
        for (int it = 0; it < 2; ++it) {
            int seg = threadIdx.x + it*128;           // 256 x 16B segments
            const _Float16* gp = kb + (size_t)(j*32)*DD + seg*8;
            _Float16*       lp = kbuf + seg*8;
#  if defined(HAVE_ASYNC_LDS)
            __builtin_amdgcn_global_load_async_to_lds_b128(as_global(gp), as_lds(lp), 0, 0);
#  else
            *(uint4*)lp = *(const uint4*)gp;
#  endif
        }
#endif
        // ---- stage V chunk (32 keys x 64 dims) transposed into LDS ----
        #pragma unroll
        for (int it = 0; it < 2; ++it) {
            int task = threadIdx.x + it*128;          // 256 tasks: (key, octet-of-8-dims)
            int key = task >> 3, oct = task & 7;
            H8 w; w.u = *(const uint4*)(vb + ((size_t)(j*32 + key))*DD + oct*8);
            #pragma unroll
            for (int e = 0; e < 8; ++e) vbufT[(oct*8 + e)*32 + key] = w.h[e];
        }
        if (j + 1 < NN/32)
            __builtin_prefetch(vb + (size_t)(j+1)*32*DD + threadIdx.x*16, 0, 0);
#if !defined(HAVE_TDM) && defined(HAVE_ASYNC_LDS)
#  if __has_builtin(__builtin_amdgcn_s_wait_asynccnt)
        __builtin_amdgcn_s_wait_asynccnt(0);
#  else
        asm volatile("s_wait_asynccnt 0x0" ::: "memory");
#  endif
#endif
        __syncthreads();

        // ---- scores: two 16-key tiles, K-dim 64 split into two WMMAs each ----
        v8f z = {};
        const _Float16* kt0 = kbuf;
        const _Float16* kt1 = kbuf + 16*DD;
        v8f s0 = __builtin_amdgcn_wmma_f32_16x16x32_f16(false, aq0, false, fragB(kt0, DD, lane, 0),  (short)0, z,  false, false);
        s0     = __builtin_amdgcn_wmma_f32_16x16x32_f16(false, aq1, false, fragB(kt0, DD, lane, 32), (short)0, s0, false, false);
        v8f s1 = __builtin_amdgcn_wmma_f32_16x16x32_f16(false, aq0, false, fragB(kt1, DD, lane, 0),  (short)0, z,  false, false);
        s1     = __builtin_amdgcn_wmma_f32_16x16x32_f16(false, aq1, false, fragB(kt1, DD, lane, 32), (short)0, s1, false, false);

        // ---- online softmax per row (row = r + 8*(lane>=16), col = lane%16) ----
        int rbase = (lane & 16) ? 8 : 0;
        #pragma unroll
        for (int r = 0; r < 8; ++r) {
            float a0 = s0[r] * 0.125f;                // 1/sqrt(64)
            float a1 = s1[r] * 0.125f;
            float mx = rowred_max(fmaxf(a0, a1));
            float mnew = fmaxf(mrow[r], mx);
            float corr = __expf(mrow[r] - mnew);
            float p0 = __expf(a0 - mnew);
            float p1 = __expf(a1 - mnew);
            float rs = rowred_sum(p0 + p1);
            lrow[r] = lrow[r] * corr + rs;
            mrow[r] = mnew;
            #pragma unroll
            for (int f = 0; f < 4; ++f) acc[f][r] *= corr;
            int row = r + rbase, col = lane & 15;
            pbuf[wave][row*32 + col]      = (_Float16)p0;
            pbuf[wave][row*32 + 16 + col] = (_Float16)p1;
        }
        __syncthreads();

        // ---- A·V: P (16x32 keys) x V (32 keys x 64 dims) in four 16-col WMMAs ----
        v16h pf = fragA(&pbuf[wave][0], 32, lane, 0);
        #pragma unroll
        for (int f = 0; f < 4; ++f) {
            v16h bv = fragB(&vbufT[(f*16)*32], 32, lane, 0);
            acc[f] = __builtin_amdgcn_wmma_f32_16x16x32_f16(false, pf, false, bv, (short)0, acc[f], false, false);
        }
        __syncthreads();
    }

    // ---- epilogue: normalize and scatter to o (B,N,256) with slot head*64 + d ----
    int b = bh >> 2, head = bh & 3;
    int rbase = (lane & 16) ? 8 : 0;
    #pragma unroll
    for (int r = 0; r < 8; ++r) {
        float inv = 1.f / lrow[r];
        int qrow = q0 + r + rbase;
        #pragma unroll
        for (int f = 0; f < 4; ++f) {
            int d = f*16 + (lane & 15);
            obuf[((size_t)(b*NN + qrow))*256 + head*64 + d] = acc[f][r] * inv;
        }
    }
}

// ---------------- x += clin(o, wo) ----------------
__global__ void wo_kernel(const float* __restrict__ obuf, const float* __restrict__ wo,
                          float* __restrict__ x) {
    __shared__ float os[256];
    int tok = blockIdx.x, t = threadIdx.x;
    os[t] = obuf[(size_t)tok*256 + t];
    __syncthreads();
    int o = t >> 4, bl = t & 15, gr = GR[bl];
    float a = 0.f;
    #pragma unroll
    for (int i = 0; i < 16; ++i) a += os[i*16 + bl] * wo[(o*16+i)*5 + gr];
    x[(size_t)tok*256 + t] += a;
}

// ---------------- fused MLP: norm -> w1 -> gp -> gate -> w2 -> residual ----------------
__global__ void mlp_kernel(float* __restrict__ x, const float* __restrict__ w1,
                           const float* __restrict__ w2) {
    __shared__ float red[256];
    __shared__ float hs[256];
    __shared__ float fs[1536];   // 96 channels x 16 blades (u: 0..1023, g: 1024..1535)
    int tok = blockIdx.x, t = threadIdx.x;
    float xorig = x[(size_t)tok*256 + t];
    red[t] = xorig * xorig;
    __syncthreads();
    for (int off = 128; off; off >>= 1) { if (t < off) red[t] += red[t+off]; __syncthreads(); }
    float s = red[0] * (1.0f/16.0f);
    hs[t] = xorig * rsqrtf(s + 1e-6f);
    __syncthreads();
    // u = clin(h, w1): 64 out channels x 16 blades
    #pragma unroll
    for (int uu = 0; uu < 4; ++uu) {
        int idx = t + uu*256;
        int o = idx >> 4, bl = idx & 15, gr = GR[bl];
        float a = 0.f;
        #pragma unroll
        for (int i = 0; i < 16; ++i) a += hs[i*16 + bl] * w1[(o*16+i)*5 + gr];
        fs[idx] = a;
    }
    __syncthreads();
    // g[c] = gp(u[c], u[c+32]) for c in 0..31  (reads u region, writes g region: disjoint)
    #pragma unroll
    for (int gg = 0; gg < 2; ++gg) {
        int idx = t + gg*256;            // 512 tasks: (channel, out-blade)
        int c = idx >> 4, kidx = idx & 15;
        int km = MASK[kidx];
        float a = 0.f;
        for (int ai = 0; ai < 16; ++ai) {
            int am = MASK[ai];
            int bm = am ^ km;
            if (am & bm & 1) continue;   // e0*e0 = 0
            float sgn = (float)gp_sign(am, bm);
            a += sgn * fs[c*16 + ai] * fs[(c+32)*16 + IDXOF[bm]];
        }
        fs[1024 + idx] = a;
    }
    __syncthreads();
    // gate: f[c] *= sigmoid(f[c][0])
    if (t < 96) {
        float sc = 1.f / (1.f + __expf(-fs[t*16]));
        #pragma unroll
        for (int bl = 0; bl < 16; ++bl) fs[t*16 + bl] *= sc;
    }
    __syncthreads();
    // x += clin(gate(f), w2)
    int o = t >> 4, bl = t & 15, gr = GR[bl];
    float a = 0.f;
    for (int i = 0; i < 96; ++i) a += fs[i*16 + bl] * w2[(o*96+i)*5 + gr];
    x[(size_t)tok*256 + t] = xorig + a;
}

// ---------------- project + residual displacement ----------------
__global__ void proj_kernel(const float* __restrict__ x, const float* __restrict__ pw,
                            const float* __restrict__ pos, float* __restrict__ out) {
    int t = blockIdx.x * blockDim.x + threadIdx.x;
    if (t >= TOKENS) return;
    const float* xt = x + (size_t)t*256;
    #pragma unroll
    for (int j = 0; j < 3; ++j) {
        int bl = 2 + j;                  // grade-1 vector blades e1,e2,e3
        float a = 0.f;
        #pragma unroll
        for (int i = 0; i < 16; ++i) a += xt[i*16 + bl] * pw[i*5 + 1];
        out[t*3 + j] = pos[t*3 + j] + a;
    }
}

extern "C" void kernel_launch(void* const* d_in, const int* in_sizes, int n_in,
                              void* d_out, int out_size, void* d_ws, size_t ws_size,
                              hipStream_t stream) {
    const float* pos  = (const float*)d_in[0];
    const float* vel  = (const float*)d_in[1];
    const float* lw   = (const float*)d_in[2];
    const float* wq   = (const float*)d_in[3];
    const float* wk   = (const float*)d_in[4];
    const float* wv   = (const float*)d_in[5];
    const float* wo   = (const float*)d_in[6];
    const float* w1   = (const float*)d_in[7];
    const float* w2   = (const float*)d_in[8];
    const float* pw   = (const float*)d_in[9];
    float* out = (float*)d_out;

    char* ws = (char*)d_ws;
    float*    x    = (float*)ws;                               // 16 MB
    _Float16* qh   = (_Float16*)(ws + (size_t)16*1024*1024);   //  8 MB
    _Float16* kh   = (_Float16*)(ws + (size_t)24*1024*1024);   //  8 MB
    _Float16* vh   = (_Float16*)(ws + (size_t)32*1024*1024);   //  8 MB
    float*    obuf = (float*)   (ws + (size_t)40*1024*1024);   // 16 MB

    lift_kernel<<<(TOKENS*CC + 255)/256, 256, 0, stream>>>(pos, vel, lw, x);

    for (int l = 0; l < 2; ++l) {
        const float* wql = wq + (size_t)l*16*16*5;
        const float* wkl = wk + (size_t)l*16*16*5;
        const float* wvl = wv + (size_t)l*16*16*5;
        const float* wol = wo + (size_t)l*16*16*5;
        const float* w1l = w1 + (size_t)l*64*16*5;
        const float* w2l = w2 + (size_t)l*16*96*5;

        qkv_kernel<<<TOKENS, 256, 0, stream>>>(x, wql, wkl, wvl, qh, kh, vh);
        attn_kernel<<<dim3(BB*HH, NN/64), 128, 0, stream>>>(qh, kh, vh, obuf);
        wo_kernel<<<TOKENS, 256, 0, stream>>>(obuf, wol, x);
        mlp_kernel<<<TOKENS, 256, 0, stream>>>(x, w1l, w2l);
    }

    proj_kernel<<<(TOKENS + 255)/256, 256, 0, stream>>>(x, pw, pos, out);
}
